// PVMFormer_81346680586863
// MI455X (gfx1250) — matrix operand order repaired and checked by reading, not compiled, not run on hardware
//
#include <hip/hip_runtime.h>
#include <hip/hip_bf16.h>

typedef __attribute__((ext_vector_type(16))) _Float16 v16h;
typedef __attribute__((ext_vector_type(8)))  _Float16 v8h;
typedef __attribute__((ext_vector_type(8)))  float    v8f;

// ---------------------------------------------------------------------------
// Stage 0: weight conversion (f32 -> f16) and A = -exp(A_log)
// ---------------------------------------------------------------------------
__global__ void __launch_bounds__(256)
k_convert(const float* __restrict__ w1, const float* __restrict__ w2,
          const float* __restrict__ A_log,
          _Float16* __restrict__ w1h, _Float16* __restrict__ w2h,
          float* __restrict__ Aneg)
{
    int idx = blockIdx.x * 256 + threadIdx.x;
    if (idx < 512 * 128) {
        w1h[idx] = (_Float16)w1[idx];
        w2h[idx] = (_Float16)w2[idx];
    }
    if (idx < 512) Aneg[idx] = -__expf(A_log[idx]);
}

// ---------------------------------------------------------------------------
// Stage 1: pre-LN over C=128 + bott_in projection (128 -> 64)
// x (8,128,4096) -> z (8,4096,64)
// ---------------------------------------------------------------------------
__global__ void __launch_bounds__(128)
k_pre_ln_proj(const float* __restrict__ x, const float* __restrict__ g,
              const float* __restrict__ bta, const float* __restrict__ w,
              const float* __restrict__ bb, float* __restrict__ z)
{
    __shared__ float xs[32][129];      // padded: bank-conflict free
    __shared__ float wsm[64 * 128];
    int b  = blockIdx.x >> 7;          // 128 pixel-tiles of 32 per batch
    int p0 = (blockIdx.x & 127) * 32;
    int tid = threadIdx.x;
    for (int i = tid; i < 64 * 128; i += 128) wsm[i] = w[i];
    for (int i = tid; i < 32 * 128; i += 128) {
        int c = i >> 5, j = i & 31;
        xs[j][c] = x[((size_t)(b * 128 + c)) * 4096 + p0 + j];
    }
    __syncthreads();
    int j = tid & 31, q = tid >> 5;
    float s = 0.f, ss = 0.f;
    for (int c = 0; c < 128; c++) { float v = xs[j][c]; s += v; ss += v * v; }
    float mean = s * (1.f / 128.f);
    float inv  = rsqrtf(ss * (1.f / 128.f) - mean * mean + 1e-5f);
    __syncthreads();
    for (int c = q * 32; c < q * 32 + 32; c++)
        xs[j][c] = (xs[j][c] - mean) * inv * g[c] + bta[c];
    __syncthreads();
    size_t zb = ((size_t)b * 4096 + p0 + j) * 64;
    for (int o = q * 16; o < q * 16 + 16; o++) {
        float acc = bb[o];
        const float* wr = &wsm[o * 128];
        #pragma unroll 8
        for (int c = 0; c < 128; c++) acc += xs[j][c] * wr[c];
        z[zb + o] = acc;
    }
}

// ---------------------------------------------------------------------------
// Stage 2: mamba front: in_proj (16->64), causal depthwise conv4 + SiLU,
// x_proj (32->33), softplus(dt).  Produces zg, xc, dtv, Bv, Cv.
// Sequences n = c*8 + b, n in [0,32).
// ---------------------------------------------------------------------------
__global__ void __launch_bounds__(128)
k_mamba_front(const float* __restrict__ z, const float* __restrict__ inw,
              const float* __restrict__ convw, const float* __restrict__ convb,
              const float* __restrict__ xpw, const float* __restrict__ dtw,
              const float* __restrict__ dtb,
              float* __restrict__ zg, float* __restrict__ xcg,
              float* __restrict__ dtv, float* __restrict__ Bv,
              float* __restrict__ Cv)
{
    __shared__ float s_inw[64 * 16];
    __shared__ float s_xpw[33 * 32];
    __shared__ float s_cw[32 * 4];
    __shared__ float s_cb[32], s_dtw[32], s_dtb[32];
    __shared__ float xcpre[131][33];   // 128-tile + 3 halo, padded
    int n  = blockIdx.x >> 5;
    int t0 = (blockIdx.x & 31) * 128;
    int c = n >> 3, b = n & 7;
    int tid = threadIdx.x;
    for (int i = tid; i < 64 * 16; i += 128) s_inw[i] = inw[i];
    for (int i = tid; i < 33 * 32; i += 128) s_xpw[i] = xpw[i];
    s_cw[tid & 127] = convw[tid & 127];
    if (tid < 32) { s_cb[tid] = convb[tid]; s_dtw[tid] = dtw[tid]; s_dtb[tid] = dtb[tid]; }
    __syncthreads();

    auto do_inproj = [&](int t, int r, bool full) {
        if (t < 0) { for (int d = 0; d < 32; d++) xcpre[r][d] = 0.f; return; }
        float z16[16];
        size_t zb = ((size_t)b * 4096 + t) * 64 + c * 16;
        #pragma unroll
        for (int i = 0; i < 16; i++) z16[i] = z[zb + i];
        int omax = full ? 64 : 32;
        for (int o = 0; o < omax; o++) {
            float a = 0.f;
            #pragma unroll
            for (int i = 0; i < 16; i++) a += z16[i] * s_inw[o * 16 + i];
            if (o < 32) xcpre[r][o] = a;
            else        zg[((size_t)n * 4096 + t) * 32 + (o - 32)] = a;
        }
    };

    int t = t0 + tid;
    do_inproj(t, 3 + tid, true);
    if (tid < 3) do_inproj(t0 - 3 + tid, tid, false);
    __syncthreads();

    float xr[32];
    #pragma unroll
    for (int d = 0; d < 32; d++) {
        float a = s_cb[d];
        #pragma unroll
        for (int k = 0; k < 4; k++) a += xcpre[tid + k][d] * s_cw[d * 4 + k];
        a = a / (1.f + __expf(-a));            // SiLU
        xr[d] = a;
    }
    size_t base = (size_t)n * 4096 + t;
    for (int d = 0; d < 32; d++) xcg[base * 32 + d] = xr[d];

    float dtraw = 0.f;
    { float a = 0.f; for (int i = 0; i < 32; i++) a += xr[i] * s_xpw[i]; dtraw = a; }
    for (int jj = 1; jj <= 16; jj++) {
        float a = 0.f;
        for (int i = 0; i < 32; i++) a += xr[i] * s_xpw[jj * 32 + i];
        Bv[base * 16 + jj - 1] = a;
    }
    for (int jj = 17; jj <= 32; jj++) {
        float a = 0.f;
        for (int i = 0; i < 32; i++) a += xr[i] * s_xpw[jj * 32 + i];
        Cv[base * 16 + jj - 17] = a;
    }
    for (int d = 0; d < 32; d++) {
        float v  = dtraw * s_dtw[d] + s_dtb[d];
        float sp = (v > 20.f) ? v : __logf(1.f + __expf(v));  // softplus
        dtv[base * 32 + d] = sp;
    }
}

// ---------------------------------------------------------------------------
// Stage 3: chunked selective scan.  h_t = exp(dt*A)*h + dt*x*B;  y = sum_s h*C
// thread (d,s): s = tid&15, d = tid>>4.  64 chunks of 64 steps.
// ---------------------------------------------------------------------------
__global__ void __launch_bounds__(512)
k_scan_p1(const float* __restrict__ dtv, const float* __restrict__ xcg,
          const float* __restrict__ Bv, const float* __restrict__ Aneg,
          float* __restrict__ Aprod, float* __restrict__ Hloc)
{
    int blk = blockIdx.x; int n = blk >> 6; int k = blk & 63;
    int tid = threadIdx.x; int s = tid & 15; int d = tid >> 4;
    float A = Aneg[d * 16 + s];
    float a = 1.f, h = 0.f;
    size_t base = (size_t)n * 4096 + k * 64;
    for (int t = 0; t < 64; t++) {
        size_t bt = base + t;
        float dt = dtv[bt * 32 + d];
        float xv = xcg[bt * 32 + d];
        float Bb = Bv[bt * 16 + s];
        float dA = __expf(dt * A);
        a *= dA;
        h = dA * h + dt * xv * Bb;
    }
    size_t o = (size_t)blk * 512 + tid;
    Aprod[o] = a; Hloc[o] = h;
}

__global__ void __launch_bounds__(512)
k_scan_p2(const float* __restrict__ Aprod, const float* __restrict__ Hloc,
          float* __restrict__ Hstart)
{
    int n = blockIdx.x; int tid = threadIdx.x;
    float h = 0.f;
    for (int k = 0; k < 64; k++) {
        size_t o = ((size_t)(n * 64 + k)) * 512 + tid;
        Hstart[o] = h;
        h = Aprod[o] * h + Hloc[o];
    }
}

__global__ void __launch_bounds__(512)
k_scan_p3(const float* __restrict__ dtv, const float* __restrict__ xcg,
          const float* __restrict__ Bv, const float* __restrict__ Cv,
          const float* __restrict__ Aneg, const float* __restrict__ Hstart,
          float* __restrict__ ys)
{
    int blk = blockIdx.x; int n = blk >> 6; int k = blk & 63;
    int tid = threadIdx.x; int s = tid & 15; int d = tid >> 4;
    float A = Aneg[d * 16 + s];
    float h = Hstart[(size_t)blk * 512 + tid];
    size_t base = (size_t)n * 4096 + k * 64;
    for (int t = 0; t < 64; t++) {
        size_t bt = base + t;
        float dt = dtv[bt * 32 + d];
        float xv = xcg[bt * 32 + d];
        float Bb = Bv[bt * 16 + s];
        float Cc = Cv[bt * 16 + s];
        float dA = __expf(dt * A);
        h = dA * h + dt * xv * Bb;
        float r = h * Cc;                       // 16-lane reduction (wave32)
        r += __shfl_xor(r, 1); r += __shfl_xor(r, 2);
        r += __shfl_xor(r, 4); r += __shfl_xor(r, 8);
        if (s == 0) ys[bt * 32 + d] = r;
    }
}

// ---------------------------------------------------------------------------
// Stage 4: gate + out_proj + residual, merge chunks, post-LN(64),
// bott_out (64->128) * res_scale, out1 = 2x + token.  One token per block.
// Also writes out1t (pixel-major f16) for the WMMA fc1.
// ---------------------------------------------------------------------------
__global__ void __launch_bounds__(64)
k_merge(const float* __restrict__ ys, const float* __restrict__ xcg,
        const float* __restrict__ zg, const float* __restrict__ z,
        const float* __restrict__ Dp, const float* __restrict__ opw,
        const float* __restrict__ pg, const float* __restrict__ pb,
        const float* __restrict__ bow, const float* __restrict__ bob,
        const float* __restrict__ res_scale, const float* __restrict__ x,
        float* __restrict__ out1, _Float16* __restrict__ out1t)
{
    __shared__ float ygs[4][32];
    __shared__ float mbuf[64];
    int p = blockIdx.x & 4095; int b = blockIdx.x >> 12;
    int j = threadIdx.x;
    int d = j & 31, c0 = j >> 5;
    for (int cc = 0; cc < 2; cc++) {
        int c = c0 + cc * 2;
        int n = c * 8 + b;
        size_t base = (size_t)n * 4096 + p;
        float xv = xcg[base * 32 + d];
        float yv = ys[base * 32 + d];
        float zv = zg[base * 32 + d];
        float sig = 1.f / (1.f + __expf(-zv));
        ygs[c][d] = (yv + xv * Dp[d]) * (zv * sig);
    }
    __syncthreads();
    int c = j >> 4, jj = j & 15;
    float acc = z[((size_t)b * 4096 + p) * 64 + j];     // + zc residual
    #pragma unroll
    for (int dd = 0; dd < 32; dd++) acc += opw[jj * 32 + dd] * ygs[c][dd];
    mbuf[j] = acc;
    __syncthreads();
    float s = 0.f, ss = 0.f;
    for (int i = 0; i < 64; i++) { float v = mbuf[i]; s += v; ss += v * v; }
    float mean = s * (1.f / 64.f);
    float inv  = rsqrtf(ss * (1.f / 64.f) - mean * mean + 1e-5f);
    __syncthreads();
    mbuf[j] = (mbuf[j] - mean) * inv * pg[j] + pb[j];
    __syncthreads();
    float rs = res_scale[0];
    for (int cc = 0; cc < 2; cc++) {
        int ch = j + cc * 64;
        float a = bob[ch];
        #pragma unroll 8
        for (int i = 0; i < 64; i++) a += mbuf[i] * bow[ch * 64 + i];
        a *= rs;
        size_t xi = ((size_t)(b * 128 + ch)) * 4096 + p;
        float o = 2.f * x[xi] + a;                      // out = x + (x + token)
        out1[xi] = o;
        out1t[((size_t)b * 4096 + p) * 128 + ch] = (_Float16)o;
    }
}

// ---------------------------------------------------------------------------
// Stage 5/7: WMMA GEMM.  out[b][M][4096] = W[M][K] * X[b][p][K]^T
// Per-wave 16x16 tile, K-loop of v_wmma_f32_16x16x32_f16.
// Fragment layouts per CDNA5 ISA 7.12.2.
// ---------------------------------------------------------------------------
template <int M, int K>
__global__ void __launch_bounds__(256)
k_gemm_wmma(const _Float16* __restrict__ W, const _Float16* __restrict__ X,
            float* __restrict__ out)
{
    const int perB = (M / 128) * 256;
    int id = blockIdx.x;
    int b = id / perB; int r = id % perB;
    int oblk = r >> 8; int pt = r & 255;
    int wave = threadIdx.x >> 5; int lane = threadIdx.x & 31;
    int o0 = oblk * 128 + wave * 16;
    int p0 = pt * 16;
    int mrow  = lane & 15;
    int khiA  = (lane >> 4) * 8;    // A: K in {khi..khi+7, khi+16..khi+23}
    int koffB = (lane >> 4) * 8;    // B: K in {khi..khi+7, khi+16..khi+23}
    const _Float16* Arow = W + (size_t)(o0 + mrow) * K + khiA;
    const _Float16* Brow = X + ((size_t)b * 4096 + p0 + mrow) * K + koffB;
    v8f acc = {};
    #pragma unroll
    for (int k0 = 0; k0 < K; k0 += 32) {
        v8h a0 = *(const v8h*)(Arow + k0);
        v8h a1 = *(const v8h*)(Arow + k0 + 16);
        v8h b0 = *(const v8h*)(Brow + k0);
        v8h b1 = *(const v8h*)(Brow + k0 + 16);
        v16h av, bv;
        #pragma unroll
        for (int i = 0; i < 8; i++) {
            av[i] = a0[i]; av[i + 8] = a1[i];
            bv[i] = b0[i]; bv[i + 8] = b1[i];
        }
        acc = __builtin_amdgcn_wmma_f32_16x16x32_f16(
                  false, av, false, bv, (short)0, acc, false, false);
    }
    int orow = (lane >> 4) * 8;
    int pcol = lane & 15;
    #pragma unroll
    for (int jv = 0; jv < 8; jv++)
        out[((size_t)(b * M) + o0 + orow + jv) * 4096 + p0 + pcol] = acc[jv];
}

// ---------------------------------------------------------------------------
// GroupNorm statistics: one block per (batch, group) -> mean, invstd
// ---------------------------------------------------------------------------
__global__ void __launch_bounds__(256)
k_gn_stats(const float* __restrict__ src, float* __restrict__ stats,
           int Cper, int C)
{
    __shared__ float ssum[256], ssq[256];
    int b = blockIdx.x >> 5; int g = blockIdx.x & 31;
    int tid = threadIdx.x;
    int n = Cper * 4096;
    float s = 0.f, q = 0.f;
    for (int idx = tid; idx < n; idx += 256) {
        int ch = idx >> 12; int p = idx & 4095;
        float v = src[((size_t)(b * C + g * Cper + ch)) * 4096 + p];
        s += v; q += v * v;
    }
    ssum[tid] = s; ssq[tid] = q;
    __syncthreads();
    for (int st = 128; st > 0; st >>= 1) {
        if (tid < st) { ssum[tid] += ssum[tid + st]; ssq[tid] += ssq[tid + st]; }
        __syncthreads();
    }
    if (tid == 0) {
        float mean = ssum[0] / n;
        float var  = ssq[0] / n - mean * mean;
        stats[blockIdx.x * 2]     = mean;
        stats[blockIdx.x * 2 + 1] = rsqrtf(var + 1e-5f);
    }
}

// ---------------------------------------------------------------------------
// Stage 6: GN1 + exact GELU, with LDS 32x32 transpose to pixel-major f16
// ---------------------------------------------------------------------------
__global__ void __launch_bounds__(256)
k_gn1_gelu_t(const float* __restrict__ h1, const float* __restrict__ stats,
             const float* __restrict__ gg, const float* __restrict__ gb,
             _Float16* __restrict__ g1t)
{
    __shared__ float tile[32][33];
    int id = blockIdx.x;
    int b = id / 2048; int r = id % 2048;
    int ot = r >> 7; int ptile = r & 127;
    int o0 = ot * 32, p0 = ptile * 32;
    int tid = threadIdx.x;
    #pragma unroll
    for (int i = 0; i < 4; i++) {
        int ol = (tid >> 5) + i * 8; int pl = tid & 31;
        int o = o0 + ol;
        float v = h1[((size_t)(b * 512 + o)) * 4096 + p0 + pl];
        int g = o >> 4;
        float mean = stats[(b * 32 + g) * 2], inv = stats[(b * 32 + g) * 2 + 1];
        v = (v - mean) * inv * gg[o] + gb[o];
        v = 0.5f * v * (1.f + erff(v * 0.70710678118f));   // exact GELU
        tile[ol][pl] = v;
    }
    __syncthreads();
    #pragma unroll
    for (int i = 0; i < 4; i++) {
        int pl = (tid >> 5) + i * 8; int ol = tid & 31;
        g1t[((size_t)(b * 4096) + p0 + pl) * 512 + o0 + ol] = (_Float16)tile[ol][pl];
    }
}

// ---------------------------------------------------------------------------
// Stage 8: GN2 + final residual add -> d_out
// ---------------------------------------------------------------------------
__global__ void __launch_bounds__(256)
k_gn2_add(const float* __restrict__ h2, const float* __restrict__ stats,
          const float* __restrict__ gg, const float* __restrict__ gb,
          const float* __restrict__ out1, float* __restrict__ out)
{
    size_t idx = (size_t)blockIdx.x * 256 + threadIdx.x;
    int ch = (int)((idx >> 12) & 127);
    int b  = (int)(idx >> 19);
    int g  = ch >> 2;
    float mean = stats[(b * 32 + g) * 2], inv = stats[(b * 32 + g) * 2 + 1];
    float v = h2[idx];
    v = (v - mean) * inv * gg[ch] + gb[ch];
    out[idx] = out1[idx] + v;
}

// ---------------------------------------------------------------------------
extern "C" void kernel_launch(void* const* d_in, const int* in_sizes, int n_in,
                              void* d_out, int out_size, void* d_ws, size_t ws_size,
                              hipStream_t stream)
{
    (void)in_sizes; (void)n_in; (void)out_size; (void)ws_size;
    const float* x        = (const float*)d_in[0];
    const float* pre_g    = (const float*)d_in[1];
    const float* pre_b    = (const float*)d_in[2];
    const float* bin_w    = (const float*)d_in[3];
    const float* bin_b    = (const float*)d_in[4];
    const float* inproj_w = (const float*)d_in[5];
    const float* conv_w   = (const float*)d_in[6];
    const float* conv_b   = (const float*)d_in[7];
    const float* xproj_w  = (const float*)d_in[8];
    const float* dtp_w    = (const float*)d_in[9];
    const float* dtp_b    = (const float*)d_in[10];
    const float* A_log    = (const float*)d_in[11];
    const float* Dp       = (const float*)d_in[12];
    const float* oproj_w  = (const float*)d_in[13];
    const float* post_g   = (const float*)d_in[14];
    const float* post_b   = (const float*)d_in[15];
    const float* bout_w   = (const float*)d_in[16];
    const float* bout_b   = (const float*)d_in[17];
    const float* res_sc   = (const float*)d_in[18];
    const float* fc1_w    = (const float*)d_in[19];
    const float* gn1_g    = (const float*)d_in[20];
    const float* gn1_b    = (const float*)d_in[21];
    const float* fc2_w    = (const float*)d_in[22];
    const float* gn2_g    = (const float*)d_in[23];
    const float* gn2_b    = (const float*)d_in[24];

    float* F = (float*)d_ws;
    float* z      = F;                    // 2,097,152
    float* zg     = F + 2097152;          // 4,194,304
    float* xc     = F + 6291456;          // 4,194,304
    float* dtv    = F + 10485760;         // 4,194,304
    float* Bv     = F + 14680064;         // 2,097,152
    float* Cv     = F + 16777216;         // 2,097,152
    float* ys     = F + 18874368;         // 4,194,304
    float* out1   = F + 23068672;         // 4,194,304
    float* h1     = F + 27262976;         // 16,777,216
    float* Aprod  = h1;                   // scan temps overlap h1 (dead later)
    float* Hloc   = h1 + 1048576;
    float* Hstart = h1 + 2097152;
    float* h2     = F + 44040192;         // 4,194,304
    float* st1    = F + 48234496;         // 512
    float* st2    = F + 48235008;         // 512
    float* Aneg   = F + 48235520;         // 512
    _Float16* Hh    = (_Float16*)(F + 48236032);
    _Float16* w1h   = Hh;                 // 65,536
    _Float16* w2h   = Hh + 65536;         // 65,536
    _Float16* out1t = Hh + 131072;        // 4,194,304
    _Float16* g1t   = Hh + 131072 + 4194304; // 16,777,216

    k_convert<<<256, 256, 0, stream>>>(fc1_w, fc2_w, A_log, w1h, w2h, Aneg);
    k_pre_ln_proj<<<1024, 128, 0, stream>>>(x, pre_g, pre_b, bin_w, bin_b, z);
    k_mamba_front<<<1024, 128, 0, stream>>>(z, inproj_w, conv_w, conv_b, xproj_w,
                                            dtp_w, dtp_b, zg, xc, dtv, Bv, Cv);
    k_scan_p1<<<2048, 512, 0, stream>>>(dtv, xc, Bv, Aneg, Aprod, Hloc);
    k_scan_p2<<<32, 512, 0, stream>>>(Aprod, Hloc, Hstart);
    k_scan_p3<<<2048, 512, 0, stream>>>(dtv, xc, Bv, Cv, Aneg, Hstart, ys);
    k_merge<<<32768, 64, 0, stream>>>(ys, xc, zg, z, Dp, oproj_w, post_g, post_b,
                                      bout_w, bout_b, res_sc, x, out1, out1t);
    k_gemm_wmma<512, 128><<<8192, 256, 0, stream>>>(w1h, out1t, h1);
    k_gn_stats<<<256, 256, 0, stream>>>(h1, st1, 16, 512);
    k_gn1_gelu_t<<<16384, 256, 0, stream>>>(h1, st1, gn1_g, gn1_b, g1t);
    k_gemm_wmma<128, 512><<<2048, 256, 0, stream>>>(w2h, g1t, h2);
    k_gn_stats<<<256, 256, 0, stream>>>(h2, st2, 4, 128);
    k_gn2_add<<<16384, 256, 0, stream>>>(h2, st2, gn2_g, gn2_b, out1, (float*)d_out);
}
